// PlasticityModelMoE_48799418417359
// MI455X (gfx1250) — compile-verified, matrix operand order checked by reading, not run on hardware
//
#include <hip/hip_runtime.h>
#include <hip/hip_bf16.h>

typedef __attribute__((ext_vector_type(16))) __bf16 v16bf;
typedef __attribute__((ext_vector_type(8)))  float  v8f;

#define B_DIM   16384
#define D_DIM   512
#define U_DIM   1024
#define K_DEN   4
#define M_MEM   4096
#define DM_DIM  512
#define NK      (U_DIM * K_DEN)   // 4096
#define WRITE_IDX 123

#define BM 128
#define BN 128
#define BK 32
#define SSTR 40   // LDS row stride (bf16 elems); 80B rows stay 16B aligned, bank-skewed

// ---------- scalar helpers ----------
__device__ __forceinline__ unsigned short f2bf(float x) {
    unsigned int u = __float_as_uint(x);
    unsigned int r = u + 0x7FFFu + ((u >> 16) & 1u);
    return (unsigned short)(r >> 16);
}
__device__ __forceinline__ float bf2f(unsigned short h) {
    return __uint_as_float(((unsigned int)h) << 16);
}
__device__ __forceinline__ float sigmoidf_(float x) { return 1.f / (1.f + expf(-x)); }

__device__ __forceinline__ void softmax9(const float* __restrict__ w9, float wt[8]) {
    float m = w9[0];
#pragma unroll
    for (int i = 1; i < 9; ++i) m = fmaxf(m, w9[i]);
    float e[9], s = 0.f;
#pragma unroll
    for (int i = 0; i < 9; ++i) { e[i] = expf(w9[i] - m); s += e[i]; }
    float inv = 1.f / s;
#pragma unroll
    for (int i = 0; i < 8; ++i) wt[i] = e[i] * inv;
}

__device__ __forceinline__ float lqa_act(float x, const float wt[8]) {
    float sig = sigmoidf_(x);
    float elu = (x > 0.f) ? x : (expf(x) - 1.f);
    float th  = tanhf(x);
    float rel = fmaxf(x, 0.f);
    float sil = x * sig;
    float gel = 0.5f * x * (1.f + erff(x * 0.70710678118654752f));
    float sel = 1.0507009873554805f * ((x > 0.f) ? x : 1.6732632423543772f * (expf(x) - 1.f));
    float sp  = (x > 20.f) ? x : log1pf(expf(x));
    float mish = x * tanhf(sp);
    return wt[0]*sig + wt[1]*elu + wt[2]*th + wt[3]*rel
         + wt[4]*sil + wt[5]*gel + wt[6]*sel + wt[7]*mish;
}

// ---------- elementwise prep kernels ----------
__global__ void cast_bf16_kernel(const float* __restrict__ src,
                                 unsigned short* __restrict__ dst, int n) {
    int i = blockIdx.x * blockDim.x + threadIdx.x;
    if (i < n) dst[i] = f2bf(src[i]);
}

// dst[n*K + k] = bf16(src[k*N + n])  (store transposed for GEMM B-operands)
__global__ void cast_bf16_t_kernel(const float* __restrict__ src,
                                   unsigned short* __restrict__ dst, int K, int N) {
    int i = blockIdx.x * blockDim.x + threadIdx.x;
    if (i < K * N) {
        int k = i / N, n = i - k * N;
        dst[(size_t)n * K + k] = f2bf(src[i]);
    }
}

// transposed w_mod: dst[n*D + d] = bf16(w[d*NK+n] * sigmoid(delay[d*NK+n]))
__global__ void wmod_t_kernel(const float* __restrict__ w, const float* __restrict__ delay,
                              unsigned short* __restrict__ dst) {
    int i = blockIdx.x * blockDim.x + threadIdx.x;
    if (i < D_DIM * NK) {
        int d = i / NK, n = i - d * NK;
        dst[(size_t)n * D_DIM + d] = f2bf(w[i] * sigmoidf_(delay[i]));
    }
}

__global__ void zero_kernel(float* __restrict__ p, int n) {
    int i = blockIdx.x * blockDim.x + threadIdx.x;
    if (i < n) p[i] = 0.f;
}

// ---------- DynamicConnectivity (tiny net, one block) ----------
__global__ void conn_kernel(const float* __restrict__ neuron_avg,
                            const float* __restrict__ dc_w1, const float* __restrict__ dc_b1,
                            const float* __restrict__ dc_w2, const float* __restrict__ dc_b2,
                            const float* __restrict__ act_w_dc,
                            float* __restrict__ conn) {
    __shared__ float h[32];
    int t = threadIdx.x;
    float wt[8];
    softmax9(act_w_dc, wt);
    if (t < 32) {
        float acc = dc_b1[t];
        for (int u = 0; u < U_DIM; ++u) acc += neuron_avg[u] * dc_w1[u * 32 + t];
        h[t] = lqa_act(acc, wt);
    }
    __syncthreads();
    for (int u = t; u < U_DIM; u += blockDim.x) {
        float s = dc_b2[u];
#pragma unroll
        for (int j = 0; j < 32; ++j) s += h[j] * dc_w2[j * U_DIM + u];
        conn[u] = sigmoidf_(s);
    }
}

// ---------- gate = softmax(x @ gate_w + gate_b): one block per batch row ----------
__global__ void gate_kernel(const float* __restrict__ x, const float* __restrict__ gate_w,
                            const float* __restrict__ gate_b, float* __restrict__ gate) {
    __shared__ float red[4][128];
    int b = blockIdx.x, t = threadIdx.x;
    float acc[4] = {0.f, 0.f, 0.f, 0.f};
    const float* xr = x + (size_t)b * D_DIM;
    for (int d = t; d < D_DIM; d += 128) {
        float xv = xr[d];
#pragma unroll
        for (int k = 0; k < 4; ++k) acc[k] += xv * gate_w[d * 4 + k];
    }
#pragma unroll
    for (int k = 0; k < 4; ++k) red[k][t] = acc[k];
    __syncthreads();
    for (int o = 64; o > 0; o >>= 1) {
        if (t < o) {
#pragma unroll
            for (int k = 0; k < 4; ++k) red[k][t] += red[k][t + o];
        }
        __syncthreads();
    }
    if (t == 0) {
        float l[4], mx = -3.4e38f, s = 0.f;
#pragma unroll
        for (int k = 0; k < 4; ++k) { l[k] = red[k][0] + gate_b[k]; mx = fmaxf(mx, l[k]); }
#pragma unroll
        for (int k = 0; k < 4; ++k) { l[k] = expf(l[k] - mx); s += l[k]; }
        float inv = 1.f / s;
#pragma unroll
        for (int k = 0; k < 4; ++k) gate[b * 4 + k] = l[k] * inv;
    }
}

// ---------- async 16B global -> LDS copy (ASYNCcnt path, CDNA5) ----------
__device__ __forceinline__ void async_b128(const unsigned short* g, unsigned short* l) {
    unsigned long long ga = (unsigned long long)(uintptr_t)g;
    unsigned       la = (unsigned)(uintptr_t)l;   // low 32b of flat LDS addr = LDS offset
    asm volatile("global_load_async_to_lds_b128 %0, %1, off" :: "v"(la), "v"(ga) : "memory");
}
__device__ __forceinline__ void wait_async0() {
    asm volatile("s_wait_asynccnt 0x0" ::: "memory");
}

// stage one 128x32 A slice + one 128x32 Bt slice (4 async b128 per thread)
__device__ __forceinline__ void stage_async(
    const unsigned short* __restrict__ Ag, int lda,
    const unsigned short* __restrict__ Bt, int ldbt,
    int m0, int n0, int kt,
    unsigned short* sA, unsigned short* sB, int tid) {
#pragma unroll
    for (int i = 0; i < 2; ++i) {
        int lin = tid + i * 256;
        int row = lin >> 2;
        int seg = lin & 3;
        async_b128(Ag + (size_t)(m0 + row) * lda + kt + seg * 8, sA + row * SSTR + seg * 8);
        async_b128(Bt + (size_t)(n0 + row) * ldbt + kt + seg * 8, sB + row * SSTR + seg * 8);
    }
}

// ---------- WMMA main loop: both operands k-contiguous; double-buffered async LDS ----------
__device__ __forceinline__ void wmma_mainloop(
    const unsigned short* __restrict__ Ag, int lda,
    const unsigned short* __restrict__ Bt, int ldbt,   // B transposed: [N][Kdim]
    int m0, int n0, int Kdim,
    unsigned short* smem, v8f acc[2][4]) {

    unsigned short* sA0 = smem;
    unsigned short* sB0 = smem + BM * SSTR;
    unsigned short* sA1 = smem + 2 * BM * SSTR;
    unsigned short* sB1 = smem + 3 * BM * SSTR;

    const int tid   = threadIdx.x;
    const int lane  = tid & 31;
    const int wid   = tid >> 5;
    const int wm    = wid >> 1;     // 0..3  (M)
    const int wn    = wid & 1;      // 0..1  (N)
    const int lr    = lane & 15;
    const int khalf = lane >> 4;

    stage_async(Ag, lda, Bt, ldbt, m0, n0, 0, sA0, sB0, tid);

    int buf = 0;
    for (int kt = 0; kt < Kdim; kt += BK, buf ^= 1) {
        wait_async0();          // our copies for current buffer are done
        __syncthreads();        // everyone's copies done; previous reads retired
        unsigned short* cA = buf ? sA1 : sA0;
        unsigned short* cB = buf ? sB1 : sB0;
        if (kt + BK < Kdim)
            stage_async(Ag, lda, Bt, ldbt, m0, n0, kt + BK,
                        buf ? sA0 : sA1, buf ? sB0 : sB1, tid);

        union FragU { v16bf v; unsigned int u[8]; };
        FragU af[2], bfr[4];
#pragma unroll
        for (int mi = 0; mi < 2; ++mi) {
            int row = wm * 32 + mi * 16 + lr;
#pragma unroll
            for (int j = 0; j < 8; ++j) {
                int k = khalf * 8 + 2 * j + ((j >= 4) ? 8 : 0);
                af[mi].u[j] = *(const unsigned int*)(cA + row * SSTR + k);
            }
        }
#pragma unroll
        for (int ni = 0; ni < 4; ++ni) {
            int col = wn * 64 + ni * 16 + lr;
#pragma unroll
            for (int j = 0; j < 8; ++j) {
                int k = khalf * 8 + 2 * j + ((j >= 4) ? 8 : 0);
                bfr[ni].u[j] = *(const unsigned int*)(cB + col * SSTR + k);
            }
        }
#pragma unroll
        for (int mi = 0; mi < 2; ++mi)
#pragma unroll
            for (int ni = 0; ni < 4; ++ni)
                acc[mi][ni] = __builtin_amdgcn_wmma_f32_16x16x32_bf16(
                    false, af[mi].v, false, bfr[ni].v, (short)0, acc[mi][ni], false, false);
    }
}

#define GEMM_PROLOGUE()                                              \
    __shared__ unsigned short smem[4 * BM * SSTR];                   \
    v8f acc[2][4];                                                   \
    {                                                                \
        v8f zz = {0.f,0.f,0.f,0.f,0.f,0.f,0.f,0.f};                  \
        for (int mi = 0; mi < 2; ++mi)                               \
            for (int ni = 0; ni < 4; ++ni) acc[mi][ni] = zz;         \
    }                                                                \
    const int m0 = blockIdx.x * BM, n0 = blockIdx.y * BN;

// ---------- GEMM1: z = (x@w_mod + b) contracted with gate, then conn/mask/LQA -> a(bf16) ----------
__global__ __launch_bounds__(256) void gemm_z_kernel(
    const unsigned short* __restrict__ xb, const unsigned short* __restrict__ wmodT,
    const float* __restrict__ gate, const float* __restrict__ bflat,
    const float* __restrict__ conn, const float* __restrict__ mask,
    const float* __restrict__ act_w_main, unsigned short* __restrict__ aout) {
    GEMM_PROLOGUE();
    wmma_mainloop(xb, D_DIM, wmodT, D_DIM, m0, n0, D_DIM, smem, acc);

    float wt[8];
    softmax9(act_w_main, wt);

    const int lane = threadIdx.x & 31, wid = threadIdx.x >> 5;
    const int wm = wid >> 1, wn = wid & 1, lr = lane & 15, khalf = lane >> 4;
#pragma unroll
    for (int mi = 0; mi < 2; ++mi) {
#pragma unroll
        for (int ni = 0; ni < 4; ++ni) {
            int n = n0 + wn * 64 + ni * 16 + lr;
            int u = n >> 2, kk = n & 3;
            float bias = bflat[n];
            float cu = conn[u] * mask[u];
#pragma unroll
            for (int r = 0; r < 8; ++r) {
                int m = m0 + wm * 32 + mi * 16 + khalf * 8 + r;
                float val = (acc[mi][ni][r] + bias) * gate[m * 4 + kk];
                float s = val + __shfl_xor(val, 1);
                s = s + __shfl_xor(s, 2);
                if (kk == 0) {
                    float zi = s * cu;
                    aout[(size_t)m * U_DIM + u] = f2bf(lqa_act(zi, wt));
                }
            }
        }
    }
}

// ---------- GEMM2: logits(bf16) = a @ em_read_w + b ----------
__global__ __launch_bounds__(256) void gemm_logits_kernel(
    const unsigned short* __restrict__ ab, const unsigned short* __restrict__ erwT,
    const float* __restrict__ bias, unsigned short* __restrict__ logits) {
    GEMM_PROLOGUE();
    wmma_mainloop(ab, U_DIM, erwT, U_DIM, m0, n0, U_DIM, smem, acc);

    const int lane = threadIdx.x & 31, wid = threadIdx.x >> 5;
    const int wm = wid >> 1, wn = wid & 1, lr = lane & 15, khalf = lane >> 4;
#pragma unroll
    for (int mi = 0; mi < 2; ++mi)
#pragma unroll
        for (int ni = 0; ni < 4; ++ni) {
            int n = n0 + wn * 64 + ni * 16 + lr;
            float bi = bias[n];
#pragma unroll
            for (int r = 0; r < 8; ++r) {
                int m = m0 + wm * 32 + mi * 16 + khalf * 8 + r;
                logits[(size_t)m * M_MEM + n] = f2bf(acc[mi][ni][r] + bi);
            }
        }
}

// ---------- row softmax over M=4096 (two-pass, in-place bf16) ----------
__global__ void rowstats_kernel(const unsigned short* __restrict__ logits,
                                float* __restrict__ stats) {
    __shared__ float red[256];
    int row = blockIdx.x, t = threadIdx.x;
    const unsigned short* lp = logits + (size_t)row * M_MEM;
    float mx = -3.4e38f;
    for (int i = t; i < M_MEM; i += 256) mx = fmaxf(mx, bf2f(lp[i]));
    red[t] = mx; __syncthreads();
    for (int o = 128; o > 0; o >>= 1) { if (t < o) red[t] = fmaxf(red[t], red[t + o]); __syncthreads(); }
    mx = red[0]; __syncthreads();
    float s = 0.f;
    for (int i = t; i < M_MEM; i += 256) s += expf(bf2f(lp[i]) - mx);
    red[t] = s; __syncthreads();
    for (int o = 128; o > 0; o >>= 1) { if (t < o) red[t] += red[t + o]; __syncthreads(); }
    if (t == 0) { stats[2 * row] = mx; stats[2 * row + 1] = 1.f / red[0]; }
}

__global__ void softmax_norm_kernel(unsigned short* __restrict__ logits,
                                    const float* __restrict__ stats, size_t total) {
    size_t stride = (size_t)gridDim.x * blockDim.x;
    for (size_t i = (size_t)blockIdx.x * blockDim.x + threadIdx.x; i < total; i += stride) {
        size_t row = i >> 12;  // /4096
        float v = expf(bf2f(logits[i]) - stats[2 * row]) * stats[2 * row + 1];
        logits[i] = f2bf(v);
    }
}

// ---------- GEMM3: read_vector(f32) = attention @ memory ----------
__global__ __launch_bounds__(256) void gemm_read_kernel(
    const unsigned short* __restrict__ att, const unsigned short* __restrict__ memT,
    float* __restrict__ out) {
    GEMM_PROLOGUE();
    wmma_mainloop(att, M_MEM, memT, M_MEM, m0, n0, M_MEM, smem, acc);

    const int lane = threadIdx.x & 31, wid = threadIdx.x >> 5;
    const int wm = wid >> 1, wn = wid & 1, lr = lane & 15, khalf = lane >> 4;
#pragma unroll
    for (int mi = 0; mi < 2; ++mi)
#pragma unroll
        for (int ni = 0; ni < 4; ++ni) {
            int n = n0 + wn * 64 + ni * 16 + lr;
#pragma unroll
            for (int r = 0; r < 8; ++r) {
                int m = m0 + wm * 32 + mi * 16 + khalf * 8 + r;
                out[(size_t)m * DM_DIM + n] = acc[mi][ni][r];
            }
        }
}

// ---------- GEMM4: column-mean of tanh(a @ em_write_w + b) into agg ----------
__global__ __launch_bounds__(256) void gemm_write_kernel(
    const unsigned short* __restrict__ ab, const unsigned short* __restrict__ ewwT,
    const float* __restrict__ bias, float* __restrict__ agg) {
    GEMM_PROLOGUE();
    wmma_mainloop(ab, U_DIM, ewwT, U_DIM, m0, n0, U_DIM, smem, acc);

    const int lane = threadIdx.x & 31, wid = threadIdx.x >> 5;
    const int wn = wid & 1, lr = lane & 15;
#pragma unroll
    for (int ni = 0; ni < 4; ++ni) {
        int n = n0 + wn * 64 + ni * 16 + lr;
        float bi = bias[n];
        float cs = 0.f;
#pragma unroll
        for (int mi = 0; mi < 2; ++mi)
#pragma unroll
            for (int r = 0; r < 8; ++r) cs += tanhf(acc[mi][ni][r] + bi);
        cs += __shfl_down(cs, 16);
        if (lane < 16) atomicAdd(&agg[n], cs);
    }
}

// ---------- new_memory = memory with row WRITE_IDX <- agg / B ----------
__global__ void finalize_memory_kernel(const float* __restrict__ memory,
                                       const float* __restrict__ agg,
                                       float* __restrict__ out, int n) {
    int i = blockIdx.x * blockDim.x + threadIdx.x;
    if (i < n) {
        int row = i / DM_DIM;
        out[i] = (row == WRITE_IDX) ? agg[i % DM_DIM] * (1.f / (float)B_DIM) : memory[i];
    }
}

// ---------- host entry ----------
extern "C" void kernel_launch(void* const* d_in, const int* in_sizes, int n_in,
                              void* d_out, int out_size, void* d_ws, size_t ws_size,
                              hipStream_t stream) {
    const float* x          = (const float*)d_in[0];
    const float* w          = (const float*)d_in[1];
    const float* bflat      = (const float*)d_in[2];
    const float* delay      = (const float*)d_in[3];
    const float* gate_w     = (const float*)d_in[4];
    const float* gate_b     = (const float*)d_in[5];
    const float* act_w_main = (const float*)d_in[6];
    const float* act_w_dc   = (const float*)d_in[7];
    const float* dc_w1      = (const float*)d_in[8];
    const float* dc_b1      = (const float*)d_in[9];
    const float* dc_w2      = (const float*)d_in[10];
    const float* dc_b2      = (const float*)d_in[11];
    const float* neuron_avg = (const float*)d_in[12];
    const float* mask       = (const float*)d_in[13];
    const float* em_read_w  = (const float*)d_in[14];
    const float* em_read_b  = (const float*)d_in[15];
    const float* em_write_w = (const float*)d_in[16];
    const float* em_write_b = (const float*)d_in[17];
    const float* memory     = (const float*)d_in[18];

    char* wsp = (char*)d_ws;
    auto alloc = [&](size_t bytes) -> void* {
        void* p = (void*)wsp;
        wsp += (bytes + 255) & ~(size_t)255;
        return p;
    };
    unsigned short* xb     = (unsigned short*)alloc((size_t)B_DIM * D_DIM * 2);
    unsigned short* wmodT  = (unsigned short*)alloc((size_t)D_DIM * NK * 2);     // [NK][D]
    unsigned short* erwT   = (unsigned short*)alloc((size_t)U_DIM * M_MEM * 2);  // [M][U]
    unsigned short* ewwT   = (unsigned short*)alloc((size_t)U_DIM * DM_DIM * 2); // [Dm][U]
    unsigned short* memT   = (unsigned short*)alloc((size_t)M_MEM * DM_DIM * 2); // [Dm][M]
    unsigned short* a_bf   = (unsigned short*)alloc((size_t)B_DIM * U_DIM * 2);
    unsigned short* logits = (unsigned short*)alloc((size_t)B_DIM * M_MEM * 2);
    float* gate  = (float*)alloc((size_t)B_DIM * 4 * 4);
    float* conn  = (float*)alloc((size_t)U_DIM * 4);
    float* stats = (float*)alloc((size_t)B_DIM * 2 * 4);
    float* agg   = (float*)alloc((size_t)DM_DIM * 4);

    float* out_read = (float*)d_out;                               // [B, Dm]
    float* out_mem  = (float*)d_out + (size_t)B_DIM * DM_DIM;      // [M, Dm]

    // prep / casts (B operands stored transposed)
    {
        int n;
        n = B_DIM * D_DIM;   cast_bf16_kernel<<<(n + 255) / 256, 256, 0, stream>>>(x, xb, n);
        n = D_DIM * NK;      wmod_t_kernel<<<(n + 255) / 256, 256, 0, stream>>>(w, delay, wmodT);
        n = U_DIM * M_MEM;   cast_bf16_t_kernel<<<(n + 255) / 256, 256, 0, stream>>>(em_read_w, erwT, U_DIM, M_MEM);
        n = U_DIM * DM_DIM;  cast_bf16_t_kernel<<<(n + 255) / 256, 256, 0, stream>>>(em_write_w, ewwT, U_DIM, DM_DIM);
        n = M_MEM * DM_DIM;  cast_bf16_t_kernel<<<(n + 255) / 256, 256, 0, stream>>>(memory, memT, M_MEM, DM_DIM);
        zero_kernel<<<(DM_DIM + 255) / 256, 256, 0, stream>>>(agg, DM_DIM);
    }
    conn_kernel<<<1, 128, 0, stream>>>(neuron_avg, dc_w1, dc_b1, dc_w2, dc_b2, act_w_dc, conn);
    gate_kernel<<<B_DIM, 128, 0, stream>>>(x, gate_w, gate_b, gate);

    gemm_z_kernel<<<dim3(B_DIM / BM, NK / BN), 256, 0, stream>>>(
        xb, wmodT, gate, bflat, conn, mask, act_w_main, a_bf);

    gemm_logits_kernel<<<dim3(B_DIM / BM, M_MEM / BN), 256, 0, stream>>>(
        a_bf, erwT, em_read_b, logits);

    rowstats_kernel<<<B_DIM, 256, 0, stream>>>(logits, stats);
    {
        size_t total = (size_t)B_DIM * M_MEM;
        int blocks = (int)((total + 256 * 8 - 1) / (256 * 8));
        softmax_norm_kernel<<<blocks, 256, 0, stream>>>(logits, stats, total);
    }

    gemm_read_kernel<<<dim3(B_DIM / BM, DM_DIM / BN), 256, 0, stream>>>(logits, memT, out_read);

    gemm_write_kernel<<<dim3(B_DIM / BM, DM_DIM / BN), 256, 0, stream>>>(
        a_bf, ewwT, em_write_b, agg);

    {
        int n = M_MEM * DM_DIM;
        finalize_memory_kernel<<<(n + 255) / 256, 256, 0, stream>>>(memory, agg, out_mem, n);
    }
    (void)in_sizes; (void)n_in; (void)out_size; (void)ws_size;
}
